// AttentionBlock_42923903156309
// MI455X (gfx1250) — compile-verified, matrix operand order checked by reading, not run on hardware
//
#include <hip/hip_runtime.h>
#include <hip/hip_bf16.h>

#define CNUM 512
#define NHEADS 8
#define DHEAD 64
#define NGROUPS 8
#define NSPATIAL 1024
#define BATCH 16
#define EPSV 1e-5f

typedef _Float16 half_t;
typedef _Float16 v16h __attribute__((ext_vector_type(16)));
typedef _Float16 v8h  __attribute__((ext_vector_type(8)));
typedef _Float16 v4h  __attribute__((ext_vector_type(4)));
typedef float    v8f  __attribute__((ext_vector_type(8)));
typedef int      v4i  __attribute__((ext_vector_type(4)));

// CDNA5 async global->LDS copy (ASYNCcnt-tracked), guarded so compilation
// succeeds on toolchains without the builtin.  Probe result: the builtin takes
// generic int4-vector pointers: (v4i* gsrc, v4i* lds_dst, imm offset, imm cpol).
#if defined(__AMDGCN__) && __has_builtin(__builtin_amdgcn_global_load_async_to_lds_b128)
#define HAS_ASYNC_LDS 1
#else
#define HAS_ASYNC_LDS 0
#endif

__device__ __forceinline__ void wait_asynccnt0() {
#if HAS_ASYNC_LDS
#if __has_builtin(__builtin_amdgcn_s_wait_asynccnt)
    __builtin_amdgcn_s_wait_asynccnt(0);
#else
    asm volatile("s_wait_asynccnt 0" ::: "memory");
#endif
#endif
}

// Load one WMMA 16-bit A/B fragment for a 32-wide K segment.
// Per CDNA5 ISA layout: lanes 0-15 hold K=[0..7]+[16..23], lanes 16-31 hold
// K=[8..15]+[24..31] (khalf = lane>>4). p must be 16B aligned.
__device__ __forceinline__ v16h ld_frag(const half_t* p, int khalf) {
    union { v16h v; v8h h[2]; } u;
    u.h[0] = *(const v8h*)(p + khalf * 8);
    u.h[1] = *(const v8h*)(p + khalf * 8 + 16);
    return u.v;
}

// ---------------------------------------------------------------------------
// Kernel 0: one-time weight conversion fp32 -> f16 (weights then live in L2)
// ---------------------------------------------------------------------------
__global__ void __launch_bounds__(256)
wconv_kernel(const float* __restrict__ src, half_t* __restrict__ dst) {
    int i = (blockIdx.x * 256 + threadIdx.x) * 4;
    float4 v = *(const float4*)(src + i);
    v4h o = { (half_t)v.x, (half_t)v.y, (half_t)v.z, (half_t)v.w };
    *(v4h*)(dst + i) = o;
}

// ---------------------------------------------------------------------------
// Kernel 1: GroupNorm  (fp32 -> f16 normalized activations, layout [b][c][n])
// ---------------------------------------------------------------------------
__global__ void __launch_bounds__(256)
gn_kernel(const float* __restrict__ x, const float* __restrict__ gsc,
          const float* __restrict__ gbi, half_t* __restrict__ xn) {
    const int bg = blockIdx.x;            // B*GROUPS blocks
    const int b = bg >> 3, g = bg & 7;
    const int CH = CNUM / NGROUPS;        // 64 channels per group
    const int base = (b * CNUM + g * CH) * NSPATIAL;
    const int tid = threadIdx.x;
    const int TOT4 = CH * NSPATIAL / 4;   // 16384 float4s

    float s = 0.f, s2 = 0.f;
    for (int i = tid; i < TOT4; i += 256) {
        float4 v = *(const float4*)(x + base + i * 4);
        s  += v.x + v.y + v.z + v.w;
        s2 += v.x * v.x + v.y * v.y + v.z * v.z + v.w * v.w;
    }
    __shared__ float red0[256];
    __shared__ float red1[256];
    red0[tid] = s; red1[tid] = s2;
    __syncthreads();
    for (int off = 128; off > 0; off >>= 1) {
        if (tid < off) { red0[tid] += red0[tid + off]; red1[tid] += red1[tid + off]; }
        __syncthreads();
    }
    const float inv_n = 1.0f / (float)(CH * NSPATIAL);
    const float mean = red0[0] * inv_n;
    const float var  = red1[0] * inv_n - mean * mean;
    const float rstd = rsqrtf(var + EPSV);
    for (int i = tid; i < TOT4; i += 256) {
        int c = g * CH + (i >> 8);        // 256 float4s per channel
        float sc = gsc[c] * rstd, bi = gbi[c] - mean * sc;
        float4 v = *(const float4*)(x + base + i * 4);
        v4h o = { (half_t)(v.x * sc + bi), (half_t)(v.y * sc + bi),
                  (half_t)(v.z * sc + bi), (half_t)(v.w * sc + bi) };
        *(v4h*)(xn + base + i * 4) = o;
    }
}

// ---------------------------------------------------------------------------
// Kernel 2: QKV projection GEMM (f16 weights).
// A-frags stream directly from global f16 weights (L2-resident, K-contiguous).
// B tile (needs n<->k transpose) is software-pipelined: next tile's global
// loads issue during the current tile's WMMA phase; only the transposed
// ds_stores sit between barriers.
// mode 0: store q transposed [b][h][n][d], 1/sqrt(d)=0.125 folded, packed v8h
// mode 1: store natural [b][c][n] via LDS re-striping -> v8h coalesced stores
// ---------------------------------------------------------------------------
__global__ void __launch_bounds__(256)
qkv_gemm_kernel(const half_t* __restrict__ Wf, const float* __restrict__ bias,
                const half_t* __restrict__ xn, half_t* __restrict__ out, int mode) {
    const int b  = blockIdx.z;
    const int m0 = blockIdx.y * 128;      // output-channel tile
    const int n0 = blockIdx.x * 128;      // spatial tile
    const int tid = threadIdx.x;
    const int lane = tid & 31, wave = tid >> 5;
    const int wm = wave >> 2, wn = wave & 3;
    const int rl = lane & 15, khalf = lane >> 4;

    __shared__ union {
        half_t B[128][40];                // [n_local][k]  (transposed B tile)
        half_t o[128][136];               // output staging
    } sm;

    v8f zero = {};
    v8f acc[4][2];
#pragma unroll
    for (int i = 0; i < 4; i++)
#pragma unroll
        for (int j = 0; j < 2; j++) acc[i][j] = zero;

    const half_t* xb = xn + (size_t)b * CNUM * NSPATIAL;
    const half_t* wrow[4];
#pragma unroll
    for (int ms = 0; ms < 4; ms++)
        wrow[ms] = Wf + (size_t)(m0 + wm * 64 + ms * 16 + rl) * CNUM;

    // B-tile register prefetch (2 chunks/thread): k0 = 0
    v8h breg[2];
#pragma unroll
    for (int t = 0; t < 2; t++) {
        int i = tid + t * 256, r = i >> 4, cc = i & 15;
        breg[t] = *(const v8h*)(xb + (size_t)r * NSPATIAL + n0 + cc * 8);
    }

    for (int k0 = 0; k0 < CNUM; k0 += 32) {
        __syncthreads();
#pragma unroll
        for (int t = 0; t < 2; t++) {
            int i = tid + t * 256, r = i >> 4, cc = i & 15;
#pragma unroll
            for (int j = 0; j < 8; j++) sm.B[cc * 8 + j][r] = breg[t][j];
        }
        __syncthreads();
        if (k0 + 32 < CNUM) {             // prefetch next B tile during compute
#pragma unroll
            for (int t = 0; t < 2; t++) {
                int i = tid + t * 256, r = i >> 4, cc = i & 15;
                breg[t] = *(const v8h*)(xb + (size_t)(k0 + 32 + r) * NSPATIAL + n0 + cc * 8);
            }
        }

        v16h af[4], bf[2];
#pragma unroll
        for (int ms = 0; ms < 4; ms++)
            af[ms] = ld_frag(wrow[ms] + k0, khalf);
#pragma unroll
        for (int ns = 0; ns < 2; ns++)
            bf[ns] = ld_frag(&sm.B[wn * 32 + ns * 16 + rl][0], khalf);
#pragma unroll
        for (int ms = 0; ms < 4; ms++)
#pragma unroll
            for (int ns = 0; ns < 2; ns++)
                acc[ms][ns] = __builtin_amdgcn_wmma_f32_16x16x32_f16(
                    false, af[ms], false, bf[ns], (short)0, acc[ms][ns], false, false);
    }

    if (mode == 0) {
        // q: d = o&63 is contiguous over the 8 accumulator rows -> packed store
#pragma unroll
        for (int ms = 0; ms < 4; ms++)
#pragma unroll
            for (int ns = 0; ns < 2; ns++) {
                const int obase = m0 + wm * 64 + ms * 16 + 8 * khalf;
                const int ncol  = n0 + wn * 32 + ns * 16 + rl;
                const int h = obase >> 6, dbase = obase & 63;
                v8h q8;
#pragma unroll
                for (int i = 0; i < 8; i++)
                    q8[i] = (half_t)((acc[ms][ns][i] + bias[obase + i]) * 0.125f);
                *(v8h*)(out + ((size_t)(b * NHEADS + h) * NSPATIAL + ncol) * DHEAD + dbase) = q8;
            }
    } else {
        // K/V: re-stripe through LDS so global stores are n-contiguous v8h
        __syncthreads();
#pragma unroll
        for (int ms = 0; ms < 4; ms++)
#pragma unroll
            for (int ns = 0; ns < 2; ns++) {
                const int ol = wm * 64 + ms * 16 + 8 * khalf;
                const int nl = wn * 32 + ns * 16 + rl;
#pragma unroll
                for (int i = 0; i < 8; i++)
                    sm.o[ol + i][nl] = (half_t)(acc[ms][ns][i] + bias[m0 + ol + i]);
            }
        __syncthreads();
        for (int i = tid; i < 2048; i += 256) {
            int r = i >> 4, ch = i & 15;
            *(v8h*)(out + ((size_t)b * CNUM + m0 + r) * NSPATIAL + n0 + ch * 8) =
                *(v8h*)&sm.o[r][ch * 8];
        }
    }
}

// ---------------------------------------------------------------------------
// Kernel 3: flash attention per (b,h). 256 thr (8 waves), 128-query block.
// Computes S^T = K^T.Q so the probability tiles are produced directly in
// WMMA A-fragment layout (no LDS re-stripe); softmax is in-lane over 64
// values + one shfl_xor(16).  V tile staged with async global->LDS copies
// (ASYNCcnt); K tile register-prefetched across block iterations.
// ---------------------------------------------------------------------------
__global__ void __launch_bounds__(256)
attn_kernel(const half_t* __restrict__ qt, const half_t* __restrict__ kk,
            const half_t* __restrict__ vv, half_t* __restrict__ ao) {
    const int bh = blockIdx.y;            // b*8+h
    const int b = bh >> 3, h = bh & 7;
    const int qb = blockIdx.x * 128;
    const int tid = threadIdx.x, lane = tid & 31, wave = tid >> 5;
    const int rl = lane & 15, khalf = lane >> 4;

    __shared__ half_t Kt[128][72];        // [m][c]  K transposed (also out staging)
    __shared__ half_t Vs[64][136];        // [c][m]  V natural

    // Q fragments (B operand): lane = query column n, K-contiguous in qt
    const int nrow = qb + wave * 16 + rl;
    const half_t* qp = qt + ((size_t)bh * NSPATIAL + nrow) * DHEAD;
    v16h qf0 = ld_frag(qp, khalf);
    v16h qf1 = ld_frag(qp + 32, khalf);

    float mcur = -1e30f, lcur = 0.f;      // per-lane softmax state for row n
    v8f zero = {};
    v8f O[4];
#pragma unroll
    for (int t = 0; t < 4; t++) O[t] = zero;

    const half_t* kbase = kk + (size_t)(b * CNUM + h * DHEAD) * NSPATIAL;
    const half_t* vbase = vv + (size_t)(b * CNUM + h * DHEAD) * NSPATIAL;

    // K-tile register prefetch (4 chunks/thread): m0 = 0
    v8h kreg[4];
#pragma unroll
    for (int t = 0; t < 4; t++) {
        int i = tid + t * 256, c = i >> 4, mc = i & 15;
        kreg[t] = *(const v8h*)(kbase + (size_t)c * NSPATIAL + mc * 8);
    }

    for (int m0 = 0; m0 < NSPATIAL; m0 += 128) {
        __syncthreads();
#pragma unroll
        for (int t = 0; t < 4; t++) {
            int i = tid + t * 256, c = i >> 4, mc = i & 15;
#if HAS_ASYNC_LDS
            __builtin_amdgcn_global_load_async_to_lds_b128(
                (v4i*)(vbase + (size_t)c * NSPATIAL + m0 + mc * 8),
                (v4i*)&Vs[c][mc * 8], 0, 0);
#else
            *(v8h*)&Vs[c][mc * 8] =
                *(const v8h*)(vbase + (size_t)c * NSPATIAL + m0 + mc * 8);
#endif
#pragma unroll
            for (int j = 0; j < 8; j++) Kt[mc * 8 + j][c] = kreg[t][j];
        }
        if (m0 + 128 < NSPATIAL) {        // prefetch next K tile during compute
#pragma unroll
            for (int t = 0; t < 4; t++) {
                int i = tid + t * 256, c = i >> 4, mc = i & 15;
                kreg[t] = *(const v8h*)(kbase + (size_t)c * NSPATIAL + m0 + 128 + mc * 8);
            }
        }
        wait_asynccnt0();
        __syncthreads();

        // ---- S^T tiles: A = K^T rows (m), B = Q columns (n) ----
        v8f S[8];
#pragma unroll
        for (int nt = 0; nt < 8; nt++) {
            v16h kf0 = ld_frag(&Kt[nt * 16 + rl][0], khalf);
            v16h kf1 = ld_frag(&Kt[nt * 16 + rl][32], khalf);
            v8f s = zero;
            s = __builtin_amdgcn_wmma_f32_16x16x32_f16(false, kf0, false, qf0, (short)0, s, false, false);
            s = __builtin_amdgcn_wmma_f32_16x16x32_f16(false, kf1, false, qf1, (short)0, s, false, false);
            S[nt] = s;
        }

        // ---- in-lane online softmax (row n = rl; keys split across lane pair) ----
        float mx = -1e30f;
#pragma unroll
        for (int nt = 0; nt < 8; nt++)
#pragma unroll
            for (int i = 0; i < 8; i++) mx = fmaxf(mx, S[nt][i]);
        mx = fmaxf(mx, __shfl_xor(mx, 16, 32));
        float mnew  = fmaxf(mcur, mx);
        float alpha = __expf(mcur - mnew);
        mcur = mnew;
        float rsum = 0.f;
#pragma unroll
        for (int nt = 0; nt < 8; nt++)
#pragma unroll
            for (int i = 0; i < 8; i++) {
                float p = __expf(S[nt][i] - mnew);
                S[nt][i] = p;
                rsum += p;
            }
        rsum += __shfl_xor(rsum, 16, 32);
        lcur = lcur * alpha + rsum;

        // O rows are n = i + 8*khalf; fetch that row's alpha from lane (i+8*khalf)
        float arow[8];
#pragma unroll
        for (int i = 0; i < 8; i++) arow[i] = __shfl(alpha, i + 8 * khalf, 32);
#pragma unroll
        for (int t = 0; t < 4; t++)
#pragma unroll
            for (int i = 0; i < 8; i++) O[t][i] *= arow[i];

        // ---- O += P @ V : P frags built directly from S^T register pairs ----
#pragma unroll
        for (int t = 0; t < 4; t++) {
            v16h pf;
#pragma unroll
            for (int j = 0; j < 8; j++) {
                pf[j]     = (half_t)S[2 * t][j];
                pf[j + 8] = (half_t)S[2 * t + 1][j];
            }
#pragma unroll
            for (int dt = 0; dt < 4; dt++) {
                v16h vf = ld_frag(&Vs[dt * 16 + rl][t * 32], khalf);
                O[dt] = __builtin_amdgcn_wmma_f32_16x16x32_f16(
                    false, pf, false, vf, (short)0, O[dt], false, false);
            }
        }
    }

    // ---- normalize (per-row 1/l via shfl), stage in Kt, coalesced v8h out ----
    float linv = 1.0f / lcur;
    float lr[8];
#pragma unroll
    for (int i = 0; i < 8; i++) lr[i] = __shfl(linv, i + 8 * khalf, 32);
    __syncthreads();
#pragma unroll
    for (int dt = 0; dt < 4; dt++)
#pragma unroll
        for (int i = 0; i < 8; i++)
            Kt[wave * 16 + i + 8 * khalf][dt * 16 + rl] = (half_t)(O[dt][i] * lr[i]);
    __syncthreads();
    for (int i = tid; i < 1024; i += 256) {
        int r = i >> 3, ch = i & 7;
        int n = qb + r;
        *(v8h*)(ao + ((size_t)b * NSPATIAL + n) * CNUM + h * DHEAD + ch * 8) =
            *(v8h*)&Kt[r][ch * 8];
    }
}

// ---------------------------------------------------------------------------
// Kernel 4: output projection + bias + residual.  Zero LDS, zero barriers:
// both operands are K-contiguous f16 in L2 (ao rows and wo rows == B^T rows),
// so fragments stream straight from global.  Residual path is v8f vectorized.
// ---------------------------------------------------------------------------
__global__ void __launch_bounds__(256)
out_gemm_kernel(const half_t* __restrict__ Wf, const float* __restrict__ bo,
                const half_t* __restrict__ ao, const float* __restrict__ x,
                float* __restrict__ y) {
    const int b  = blockIdx.z;
    const int n0 = blockIdx.y * 128;      // spatial tile (M dim)
    const int o0 = blockIdx.x * 128;      // channel tile (N dim)
    const int tid = threadIdx.x;
    const int lane = tid & 31, wave = tid >> 5;
    const int wm = wave >> 2, wn = wave & 3;
    const int rl = lane & 15, khalf = lane >> 4;

    v8f zero = {};
    v8f acc[4][2];
#pragma unroll
    for (int i = 0; i < 4; i++)
#pragma unroll
        for (int j = 0; j < 2; j++) acc[i][j] = zero;

    const half_t* arow[4];
#pragma unroll
    for (int ms = 0; ms < 4; ms++)
        arow[ms] = ao + ((size_t)b * NSPATIAL + n0 + wm * 64 + ms * 16 + rl) * CNUM;
    const half_t* brow[2];
#pragma unroll
    for (int ns = 0; ns < 2; ns++)
        brow[ns] = Wf + (size_t)(o0 + wn * 32 + ns * 16 + rl) * CNUM;

    for (int k0 = 0; k0 < CNUM; k0 += 32) {
        v16h af[4], bf[2];
#pragma unroll
        for (int ms = 0; ms < 4; ms++) af[ms] = ld_frag(arow[ms] + k0, khalf);
#pragma unroll
        for (int ns = 0; ns < 2; ns++) bf[ns] = ld_frag(brow[ns] + k0, khalf);
#pragma unroll
        for (int ms = 0; ms < 4; ms++)
#pragma unroll
            for (int ns = 0; ns < 2; ns++)
                acc[ms][ns] = __builtin_amdgcn_wmma_f32_16x16x32_f16(
                    false, af[ms], false, bf[ns], (short)0, acc[ms][ns], false, false);
    }

#pragma unroll
    for (int ms = 0; ms < 4; ms++)
#pragma unroll
        for (int ns = 0; ns < 2; ns++) {
            const int nbase = n0 + wm * 64 + ms * 16 + 8 * khalf;  // contiguous over i
            const int o     = o0 + wn * 32 + ns * 16 + rl;
            const float bb  = bo[o];
            size_t idx = ((size_t)b * CNUM + o) * NSPATIAL + nbase;
            v8f xr = *(const v8f*)(x + idx);
            v8f r  = acc[ms][ns] + xr + bb;
            *(v8f*)(y + idx) = r;
        }
}

// ---------------------------------------------------------------------------
extern "C" void kernel_launch(void* const* d_in, const int* in_sizes, int n_in,
                              void* d_out, int out_size, void* d_ws, size_t ws_size,
                              hipStream_t stream) {
    const float* x   = (const float*)d_in[0];
    const float* gsc = (const float*)d_in[1];
    const float* gbi = (const float*)d_in[2];
    const float* wq  = (const float*)d_in[3];
    const float* bq  = (const float*)d_in[4];
    const float* wk  = (const float*)d_in[5];
    const float* bk  = (const float*)d_in[6];
    const float* wv  = (const float*)d_in[7];
    const float* bv  = (const float*)d_in[8];
    const float* wo  = (const float*)d_in[9];
    const float* bo  = (const float*)d_in[10];
    float* out = (float*)d_out;

    char* ws = (char*)d_ws;
    const size_t SZ = (size_t)BATCH * CNUM * NSPATIAL * sizeof(half_t);  // 16 MB each
    const size_t WSZ = (size_t)CNUM * CNUM * sizeof(half_t);             // 512 KB each
    half_t* xn   = (half_t*)(ws + 0 * SZ);   // groupnorm output, [b][c][n]
    half_t* qtb  = (half_t*)(ws + 1 * SZ);   // q transposed+scaled, [b][h][n][d]
    half_t* kh   = (half_t*)(ws + 2 * SZ);   // k, [b][c][n]
    half_t* vh   = (half_t*)(ws + 3 * SZ);   // v, [b][c][n]
    half_t* ao   = (half_t*)(ws + 4 * SZ);   // attention out, [b][n][c]
    half_t* wq16 = (half_t*)(ws + 5 * SZ);
    half_t* wk16 = (half_t*)(ws + 5 * SZ + 1 * WSZ);
    half_t* wv16 = (half_t*)(ws + 5 * SZ + 2 * WSZ);
    half_t* wo16 = (half_t*)(ws + 5 * SZ + 3 * WSZ);

    const int WBLK = CNUM * CNUM / 4 / 256;  // 256 blocks per matrix
    wconv_kernel<<<dim3(WBLK), 256, 0, stream>>>(wq, wq16);
    wconv_kernel<<<dim3(WBLK), 256, 0, stream>>>(wk, wk16);
    wconv_kernel<<<dim3(WBLK), 256, 0, stream>>>(wv, wv16);
    wconv_kernel<<<dim3(WBLK), 256, 0, stream>>>(wo, wo16);

    gn_kernel<<<dim3(BATCH * NGROUPS), 256, 0, stream>>>(x, gsc, gbi, xn);

    qkv_gemm_kernel<<<dim3(8, 4, BATCH), 256, 0, stream>>>(wq16, bq, xn, qtb, 0);
    qkv_gemm_kernel<<<dim3(8, 4, BATCH), 256, 0, stream>>>(wk16, bk, xn, kh, 1);
    qkv_gemm_kernel<<<dim3(8, 4, BATCH), 256, 0, stream>>>(wv16, bv, xn, vh, 1);

    attn_kernel<<<dim3(NSPATIAL / 128, BATCH * NHEADS), 256, 0, stream>>>(qtb, kh, vh, ao);

    out_gemm_kernel<<<dim3(4, 8, BATCH), 256, 0, stream>>>(wo16, bo, ao, x, out);
}
